// priorDecoder_38972533243979
// MI455X (gfx1250) — 8-bit matrix operand order repaired and checked by reading, compiled, not run on hardware
//
#include <hip/hip_runtime.h>
#include <hip/hip_fp16.h>


typedef int      v8i   __attribute__((ext_vector_type(8)));
typedef unsigned u32x4 __attribute__((ext_vector_type(4)));
typedef int      i32x4 __attribute__((ext_vector_type(4)));
typedef int      i32x8 __attribute__((ext_vector_type(8)));
typedef int      vsi4  __attribute__((vector_size(16)));

#define DEV __device__ __forceinline__

DEV unsigned lds_off(const void* p) {
  return (unsigned)(size_t)(__attribute__((address_space(3))) const void*)p;
}


__global__ void k_absmax(const float* __restrict__ x, int n, unsigned* __restrict__ slot) {
  float m = 0.f;
  for (int i = blockIdx.x * blockDim.x + threadIdx.x; i < n; i += gridDim.x * blockDim.x)
    m = fmaxf(m, fabsf(x[i]));
  for (int off = 16; off > 0; off >>= 1) m = fmaxf(m, __shfl_xor(m, off, 32));
  if ((threadIdx.x & 31) == 0) atomicMax(slot, __float_as_uint(m));
}

__global__ void k_quant_input(const float* __restrict__ x, signed char* __restrict__ q,
                              const unsigned* __restrict__ slot) {
  int i = blockIdx.x * blockDim.x + threadIdx.x;
  if (i >= 4 * 64 * 64 * 128) return;
  float s = fmaxf(__uint_as_float(*slot) / 127.f, 1e-8f);
  float inv = 1.f / s;
  int c = i & 127, w = (i >> 7) & 63, h = (i >> 13) & 63, n = i >> 19;
  float v = x[((n * 128 + c) * 64 + h) * 64 + w];
  float r = fminf(fmaxf(rintf(v * inv), -127.f), 127.f);
  q[i] = (signed char)r;
}

__global__ void k_quant_act(const __half* __restrict__ h, unsigned char* __restrict__ q,
                            int n, const unsigned* __restrict__ slot) {
  int i = blockIdx.x * blockDim.x + threadIdx.x;
  if (i >= n) return;
  float s = fmaxf(__uint_as_float(*slot) / 255.f, 1e-8f);
  float inv = 1.f / s;
  float v = __half2float(h[i]);
  float r = fminf(fmaxf(rintf(v * inv), 0.f), 255.f);
  q[i] = (unsigned char)r;
}

template <int CIN, int KH, int KW, int COUT>
__global__ void k_quant_weight(const float* __restrict__ w, signed char* __restrict__ wq,
                               float* __restrict__ wscale) {
  __shared__ float red[256];
  const int oc = blockIdx.x;
  const int E = CIN * KH * KW;
  float m = 0.f;
  for (int i = threadIdx.x; i < E; i += 256) m = fmaxf(m, fabsf(w[(size_t)oc * E + i]));
  red[threadIdx.x] = m;
  __syncthreads();
  for (int o = 128; o > 0; o >>= 1) {
    if ((int)threadIdx.x < o) red[threadIdx.x] = fmaxf(red[threadIdx.x], red[threadIdx.x + o]);
    __syncthreads();
  }
  float s = fmaxf(red[0] / 127.f, 1e-8f);
  if (threadIdx.x == 0) wscale[oc] = s;
  float inv = 1.f / s;
  for (int i = threadIdx.x; i < E; i += 256) {
    int ic = i / (KH * KW), kh = (i / KW) % KH, kw = i % KW;
    float r = fminf(fmaxf(rintf(w[(size_t)oc * E + i] * inv), -127.f), 127.f);
    wq[((size_t)((kh * KW + kw) * COUT + oc)) * CIN + ic] = (signed char)r;
  }
}

DEV v8i load_a_lds(const signed char* sm, int kc) {
  int lane = threadIdx.x & 31;
  int m = lane & 15, hi = lane >> 4;
  const signed char* base = sm + m * 128 + kc * 64 + hi * 16;
  v8i a;
#pragma unroll
  for (int t = 0; t < 4; ++t) {
    int2 d = *reinterpret_cast<const int2*>(base + 32 * (t >> 1) + 8 * (t & 1));
    a[2 * t] = d.x;
    a[2 * t + 1] = d.y;
  }
  return a;
}

DEV v8i load_b_glb(const signed char* wq, int tap, int coutBase, int kc, int Cout) {
  int lane = threadIdx.x & 31;
  int nn = lane & 15, koff = (lane >> 4) << 4;
  const signed char* base = wq + ((size_t)(tap * Cout + coutBase + nn)) * 128 + kc * 64 + koff;
  int4 g0 = *reinterpret_cast<const int4*>(base);
  int4 g1 = *reinterpret_cast<const int4*>(base + 32);
  v8i b;
  b[0] = g0.x; b[1] = g0.y; b[2] = g0.z; b[3] = g0.w;
  b[4] = g1.x; b[5] = g1.y; b[6] = g1.z; b[7] = g1.w;
  return b;
}


DEV void stage_manual(signed char* smA, const signed char* __restrict__ act,
                      size_t rowBase, int ibw, int win) {
  int p = threadIdx.x >> 4;
  int boff = (threadIdx.x & 15) * 8;
  int iw = ibw + p;
  int2 v = make_int2(0, 0);
  if (iw >= 0 && iw < win)
    v = *reinterpret_cast<const int2*>(act + rowBase + (size_t)iw * 128 + boff);
  *reinterpret_cast<int2*>(smA + p * 128 + boff) = v;
}

DEV void tdm_issue_2kb(const signed char* gsrc, const signed char* smA) {
#if __has_builtin(__builtin_amdgcn_tensor_load_to_lds)
  unsigned long long ga = (unsigned long long)(size_t)gsrc;
  u32x4 g0;
  g0[0] = 1u;
  g0[1] = lds_off(smA);
  g0[2] = (unsigned)(ga & 0xffffffffu);
  g0[3] = (unsigned)((ga >> 32) & 0x1ffffffu) | (2u << 30);
  i32x8 g1;
  g1[0] = 0;
  g1[1] = (int)(2048u << 16);
  g1[2] = (int)(1u << 16);
  g1[3] = (int)(2048u << 16);
  g1[4] = 1;
  g1[5] = 2048;
  g1[6] = 0;
  g1[7] = 0;
  i32x4 z4 = {0, 0, 0, 0};
#if __clang_major__ >= 23
  i32x8 z8 = {0, 0, 0, 0, 0, 0, 0, 0};
  __builtin_amdgcn_tensor_load_to_lds(g0, g1, z4, z4, z8, 0);
#else
  __builtin_amdgcn_tensor_load_to_lds(g0, g1, z4, z4, 0);
#endif
#else
  (void)gsrc; (void)smA;
#endif
}

template <int HIN, int WIN, int COUT, bool SGNA>
__global__ void __launch_bounds__(256, 2) k_upconv(
    const signed char* __restrict__ act,
    const signed char* __restrict__ wq,
    const float* __restrict__ wscale, const float* __restrict__ bias,
    const unsigned* __restrict__ slot_in, float div_in,
    __half* __restrict__ out,
    unsigned* __restrict__ slot_out) {
  constexpr int HOUT = 2 * HIN, WOUT = 2 * WIN;
  constexpr int TPR = WIN / 8;
  __shared__ __align__(16) signed char smA[2][16 * 128];

  int tile = blockIdx.x;
  int owt = tile % TPR;
  int oh = (tile / TPR) % HOUT;
  int n = tile / (TPR * HOUT);
  int pw = owt & 1;
  int owbase = 32 * (owt >> 1) + pw;
  int ph = oh & 1;
  int wid = threadIdx.x >> 5;
  int coutBase = wid * 16;
  int lane = threadIdx.x & 31;

  int nh = 3 - ph, nw = 3 - pw;
  int nt = nh * nw;
  int ihBase = (ph == 0) ? (oh >> 1) - 1 : ((oh - 1) >> 1);

  float s_in = fmaxf(__uint_as_float(*slot_in) / div_in, 1e-8f);

  v8i acc0 = {0, 0, 0, 0, 0, 0, 0, 0};
  v8i acc1 = {0, 0, 0, 0, 0, 0, 0, 0};

  auto tapIH  = [&](int i) { return ihBase + i / nw; };
  auto tapIBW = [&](int i) { return (owbase + (2 * (i % nw) + pw) - 2) >> 1; };
  auto tapIdx = [&](int i) { return (2 * (i / nw) + ph) * 5 + (2 * (i % nw) + pw); };
  auto tapTDM = [&](int i) {
    int ih = tapIH(i), ibw = tapIBW(i);
    return (ih >= 0) && (ih < HIN) && (ibw >= 0) && (ibw + 16 <= WIN);
  };
  auto stage = [&](int i, int buf) {
    int ih = tapIH(i), ibw = tapIBW(i);
    bool rowOk = (ih >= 0) && (ih < HIN);
    size_t rowBase = ((size_t)(n * HIN + (rowOk ? ih : 0)) * WIN) * 128;
#if __has_builtin(__builtin_amdgcn_tensor_load_to_lds)
    if (rowOk && ibw >= 0 && ibw + 16 <= WIN) {
      if (wid == 0) tdm_issue_2kb(act + rowBase + (size_t)ibw * 128, smA[buf]);
    } else {
      stage_manual(smA[buf], act, rowBase, ibw, rowOk ? WIN : 0);
    }
#else
    stage_manual(smA[buf], act, rowBase, ibw, rowOk ? WIN : 0);
#endif
  };

  stage(0, 0);
#pragma unroll
  for (int i = 0; i < 9; ++i) {
    if (i >= nt) break;
    __syncthreads();
    if (i + 1 < nt) stage(i + 1, (i + 1) & 1);
#if __has_builtin(__builtin_amdgcn_tensor_load_to_lds)
    if (wid == 0) {
      if ((i + 1 < nt) && tapTDM(i + 1)) __builtin_amdgcn_s_wait_tensorcnt(1);
      else                               __builtin_amdgcn_s_wait_tensorcnt(0);
    }
#endif
    __syncthreads();
    int tap = tapIdx(i);
    __builtin_prefetch(wq + (size_t)(((i + 1 < nt) ? tapIdx(i + 1) : tap) * COUT + coutBase) * 128, 0, 3);
    const signed char* buf = smA[i & 1];
    v8i a0 = load_a_lds(buf, 0);
    v8i b0 = load_b_glb(wq, tap, coutBase, 0, COUT);
    acc0 = __builtin_amdgcn_wmma_i32_16x16x64_iu8(SGNA, a0, true, b0, acc0, false, false);
    v8i a1 = load_a_lds(buf, 1);
    v8i b1 = load_b_glb(wq, tap, coutBase, 1, COUT);
    acc1 = __builtin_amdgcn_wmma_i32_16x16x64_iu8(SGNA, a1, true, b1, acc1, false, false);
  }

  int oc = coutBase + (lane & 15);
  int hi = lane >> 4;
  float f = s_in * wscale[oc];
  float bb = bias[oc];
  float mx = 0.f;
#pragma unroll
  for (int r = 0; r < 8; ++r) {
    int m = r + hi * 8;
    int ow = owbase + 2 * m;
    float v = (float)(acc0[r] + acc1[r]) * f + bb;
    v = fmaxf(v, 0.f);
    mx = fmaxf(mx, v);
    out[((size_t)((n * HOUT + oh) * WOUT + ow)) * COUT + oc] = __float2half(v);
  }
  for (int off = 16; off > 0; off >>= 1) mx = fmaxf(mx, __shfl_xor(mx, off, 32));
  if (lane == 0) atomicMax(slot_out, __float_as_uint(mx));
}

__global__ void __launch_bounds__(256, 2) k_conv3(
    const signed char* __restrict__ act,
    const signed char* __restrict__ wq,
    const float* __restrict__ wscale, const float* __restrict__ bias,
    const unsigned* __restrict__ slot_in,
    float* __restrict__ out) {
  constexpr int H = 256, W = 256, COUT = 192;
  __shared__ __align__(16) signed char smA[16 * 128];
  int tile = blockIdx.x;
  int owt = tile % (W / 16);
  int oh = (tile / (W / 16)) % H;
  int n = tile / ((W / 16) * H);
  int owbase = owt * 16;
  int wid = threadIdx.x >> 5;
  int ct = blockIdx.y * 8 + wid;
  bool active = ct * 16 < COUT;
  int coutBase = active ? ct * 16 : 0;
  int lane = threadIdx.x & 31;
  float s_in = fmaxf(__uint_as_float(*slot_in) / 255.f, 1e-8f);
  v8i acc0 = {0, 0, 0, 0, 0, 0, 0, 0};
  v8i acc1 = {0, 0, 0, 0, 0, 0, 0, 0};
  for (int kh = 0; kh < 3; ++kh) {
    int ih = oh + kh - 1;
    if (ih < 0 || ih >= H) continue;
    for (int kw = 0; kw < 3; ++kw) {
      int ibw = owbase + kw - 1;
      int tap = kh * 3 + kw;
      size_t rowBase = ((size_t)(n * H + ih) * W) * 128;
      bool inb = (ibw >= 0) && (ibw + 16 <= W);
      __builtin_prefetch(wq + (size_t)(tap * COUT + coutBase) * 128, 0, 3);
      __syncthreads();
#if __has_builtin(__builtin_amdgcn_global_load_async_to_lds_b128)
      if (inb) {
        if (wid < 4) {
          int idx = wid * 32 + lane;
          int px = idx >> 3, bo = (idx & 7) * 16;
          const signed char* g = act + rowBase + (size_t)(ibw + px) * 128 + bo;
          __builtin_amdgcn_global_load_async_to_lds_b128(
              (vsi4*)(void*)g, (vsi4*)(void*)(smA + px * 128 + bo), 0, 0);
          asm volatile("s_wait_asynccnt 0x0" ::: "memory");
        }
      } else {
        stage_manual(smA, act, rowBase, ibw, W);
      }
#else
      stage_manual(smA, act, rowBase, ibw, W);
      (void)inb;
#endif
      __syncthreads();
      v8i a0 = load_a_lds(smA, 0);
      v8i b0 = load_b_glb(wq, tap, coutBase, 0, COUT);
      acc0 = __builtin_amdgcn_wmma_i32_16x16x64_iu8(false, a0, true, b0, acc0, false, false);
      v8i a1 = load_a_lds(smA, 1);
      v8i b1 = load_b_glb(wq, tap, coutBase, 1, COUT);
      acc1 = __builtin_amdgcn_wmma_i32_16x16x64_iu8(false, a1, true, b1, acc1, false, false);
    }
  }
  if (!active) return;
  int oc = coutBase + (lane & 15);
  int hi = lane >> 4;
  float f = s_in * wscale[oc];
  float bb = bias[oc];
#pragma unroll
  for (int r = 0; r < 8; ++r) {
    int m = r + hi * 8;
    int ow = owbase + m;
    float v = (float)(acc0[r] + acc1[r]) * f + bb;
    out[((size_t)((n * COUT + oc) * H + oh)) * W + ow] = expf(v);
  }
}


extern "C" void kernel_launch(void* const* d_in, const int* in_sizes, int n_in,
                              void* d_out, int out_size, void* d_ws, size_t ws_size,
                              hipStream_t stream) {
  (void)in_sizes; (void)n_in; (void)out_size; (void)ws_size;
  const float* x  = (const float*)d_in[0];
  const float* w1 = (const float*)d_in[1];
  const float* b1 = (const float*)d_in[2];
  const float* w2 = (const float*)d_in[3];
  const float* b2 = (const float*)d_in[4];
  const float* w3 = (const float*)d_in[5];
  const float* b3 = (const float*)d_in[6];

  char* ws = (char*)d_ws;
  size_t off = 0;
  auto carve = [&](size_t bytes) -> void* {
    void* p = ws + off;
    off += (bytes + 255) & ~(size_t)255;
    return p;
  };

  unsigned* slots = (unsigned*)carve(256);
  signed char* wq1 = (signed char*)carve((size_t)25 * 128 * 128);
  float* wsc1 = (float*)carve(128 * 4);
  signed char* wq2 = (signed char*)carve((size_t)25 * 128 * 128);
  float* wsc2 = (float*)carve(128 * 4);
  signed char* wq3 = (signed char*)carve((size_t)9 * 192 * 128);
  float* wsc3 = (float*)carve(192 * 4);
  signed char* act0 = (signed char*)carve((size_t)4 * 64 * 64 * 128);
  __half* h1 = (__half*)carve((size_t)4 * 128 * 128 * 128 * 2);
  unsigned char* act1 = (unsigned char*)carve((size_t)4 * 128 * 128 * 128);
  __half* h2 = (__half*)carve((size_t)4 * 256 * 256 * 128 * 2);
  unsigned char* act2 = (unsigned char*)carve((size_t)4 * 256 * 256 * 128);

  (void)hipMemsetAsync(slots, 0, 256, stream);

  const int NX = 4 * 128 * 64 * 64;
  k_absmax<<<1024, 256, 0, stream>>>(x, NX, slots + 0);

  k_quant_weight<128, 5, 5, 128><<<128, 256, 0, stream>>>(w1, wq1, wsc1);
  k_quant_weight<128, 5, 5, 128><<<128, 256, 0, stream>>>(w2, wq2, wsc2);
  k_quant_weight<128, 3, 3, 192><<<192, 256, 0, stream>>>(w3, wq3, wsc3);

  k_quant_input<<<(NX + 255) / 256, 256, 0, stream>>>(x, act0, slots + 0);

  k_upconv<64, 64, 128, true><<<4 * 128 * 8, 256, 0, stream>>>(
      act0, wq1, wsc1, b1, slots + 0, 127.f, h1, slots + 1);

  const int N1 = 4 * 128 * 128 * 128;
  k_quant_act<<<(N1 + 255) / 256, 256, 0, stream>>>(h1, act1, N1, slots + 1);

  k_upconv<128, 128, 128, false><<<4 * 256 * 16, 256, 0, stream>>>(
      (const signed char*)act1, wq2, wsc2, b2, slots + 1, 255.f, h2, slots + 2);

  const int N2 = 4 * 256 * 256 * 128;
  k_quant_act<<<(N2 + 255) / 256, 256, 0, stream>>>(h2, act2, N2, slots + 2);

  k_conv3<<<dim3(4 * 256 * 16, 2), 256, 0, stream>>>(
      (const signed char*)act2, wq3, wsc3, b3, slots + 2, (float*)d_out);
}
